// PatchHARv3_30288109371515
// MI455X (gfx1250) — matrix operand order repaired and checked by reading, not compiled, not run on hardware
//
#include <hip/hip_runtime.h>
#include <hip/hip_bf16.h>

// ---------------------------------------------------------------------------
// Types for CDNA5 WMMA (wave32): v_wmma_f32_16x16x32_bf16
// ---------------------------------------------------------------------------
typedef __attribute__((ext_vector_type(8)))  __bf16 v8bf;
typedef __attribute__((ext_vector_type(16))) __bf16 v16bf;
typedef __attribute__((ext_vector_type(8)))  float  v8f;

union BF16x16 { struct { v8bf lo, hi; } s; v16bf v; };

__device__ inline v8f wmma_bf16(v16bf a, v16bf b, v8f c) {
  // (neg_a, A, neg_b, B, c_mod, C, reuse_a, reuse_b)
  return __builtin_amdgcn_wmma_f32_16x16x32_bf16(false, a, false, b, (short)0, c,
                                                 false, false);
}

__device__ inline float redmax16(float v) {
#pragma unroll
  for (int m = 1; m <= 8; m <<= 1) v = fmaxf(v, __shfl_xor(v, m, 32));
  return v;
}
__device__ inline float redsum16(float v) {
#pragma unroll
  for (int m = 1; m <= 8; m <<= 1) v += __shfl_xor(v, m, 32);
  return v;
}

// Problem constants (from setup_inputs)
#define BSZ 4
#define NTOK 1024
#define DMODEL 1024
#define NH 16
#define DH 64
#define MROWS (BSZ * NTOK)   // 4096
#define PLEN 75

// ---------------------------------------------------------------------------
// Weight repacks: f32 (K,N) -> bf16 (N,K)   [B-fragment wants K-contiguous]
// ---------------------------------------------------------------------------
__global__ void k_transpose_cast(const float* __restrict__ W, __bf16* __restrict__ Wt,
                                 int K, int N) {
  int idx = blockIdx.x * 256 + threadIdx.x;
  if (idx >= K * N) return;
  int n = idx / K, k = idx - n * K;
  Wt[(size_t)n * K + k] = (__bf16)W[(size_t)k * N + n];
}

// conv weights (O,I,T) f32 -> [t][o][i] bf16
__global__ void k_repack_conv(const float* __restrict__ w, __bf16* __restrict__ Wt,
                              int O, int I, int T) {
  int idx = blockIdx.x * 256 + threadIdx.x;
  if (idx >= T * O * I) return;
  int t = idx / (O * I);
  int rem = idx - t * O * I;
  int o = rem / I, i = rem - o * I;
  Wt[idx] = (__bf16)w[((size_t)o * I + i) * T + t];
}

// ---------------------------------------------------------------------------
// Patch embed: x = patches @ embed_w + embed_b ; also write bf16 mirror xh
// ---------------------------------------------------------------------------
__global__ __launch_bounds__(256)
void k_embed(const float* __restrict__ patches, const float* __restrict__ Ew,
             const float* __restrict__ Eb, float* __restrict__ x,
             __bf16* __restrict__ xh) {
  __shared__ float p[80];
  int row = blockIdx.x;              // b*NTOK + n
  int tid = threadIdx.x;
  if (tid < PLEN) p[tid] = patches[(size_t)row * PLEN + tid];
  __syncthreads();
#pragma unroll
  for (int i = 0; i < 4; ++i) {
    int d = tid + i * 256;
    float acc = Eb[d];
    for (int k = 0; k < PLEN; ++k) acc = fmaf(p[k], Ew[(size_t)k * DMODEL + d], acc);
    x[(size_t)row * DMODEL + d] = acc;
    xh[(size_t)row * DMODEL + d] = (__bf16)acc;
  }
}

// ---------------------------------------------------------------------------
// Dense bf16 WMMA GEMM, K = 1024 fixed, M = 4096 fixed, N multiple of 128.
// Block tile 128x128, 8 waves; wave tile 32x64 => 8 wmma / k-step.
// Register-staged double-buffered LDS, ONE barrier per k-step.
// k-loop pinned to unroll 1 so register pressure stays at one iteration.
// ---------------------------------------------------------------------------
#define LDS_P 40   // padded LDS row pitch (bf16) -> conflict-free b128 reads
__global__ __launch_bounds__(256)
void k_gemm_dense(const __bf16* __restrict__ A, const __bf16* __restrict__ Wt,
                  const float* __restrict__ bias, float* __restrict__ C, int N) {
  __shared__ __bf16 As[2 * 128 * LDS_P];
  __shared__ __bf16 Bs[2 * 128 * LDS_P];
  const int tid = threadIdx.x;
  const int wave = tid >> 5, lane = tid & 31;
  const int hlf = lane >> 4, lm = lane & 15;
  const int wm = wave >> 1, wn = wave & 1;
  const int bm = blockIdx.y * 128, bn = blockIdx.x * 128;
  // staging: each thread moves 16 bf16 of A and 16 of B per k-step
  const int srow = tid >> 1, sk = (tid & 1) * 16;

  const __bf16* aptr = A + (size_t)(bm + srow) * 1024 + sk;
  const __bf16* bptr = Wt + (size_t)(bn + srow) * 1024 + sk;
  v8bf ar0 = *(const v8bf*)aptr, ar1 = *(const v8bf*)(aptr + 8);
  v8bf br0 = *(const v8bf*)bptr, br1 = *(const v8bf*)(bptr + 8);

  v8f acc[2][4];
#pragma unroll
  for (int u = 0; u < 2; ++u)
#pragma unroll
    for (int ct = 0; ct < 4; ++ct)
#pragma unroll
      for (int r = 0; r < 8; ++r) acc[u][ct][r] = 0.0f;

  // precomputed per-thread LDS element offsets
  const int stoff = srow * LDS_P + sk;
  const int a0off = (wm * 32 + lm) * LDS_P;
  const int a1off = (wm * 32 + 16 + lm) * LDS_P;
  const int bboff = (wn * 64 + lm) * LDS_P;

#pragma unroll 1
  for (int s = 0; s < 32; ++s) {
    const int bufoff = (s & 1) ? 128 * LDS_P : 0;
    *(v8bf*)(&As[bufoff + stoff])     = ar0;
    *(v8bf*)(&As[bufoff + stoff + 8]) = ar1;
    *(v8bf*)(&Bs[bufoff + stoff])     = br0;
    *(v8bf*)(&Bs[bufoff + stoff + 8]) = br1;
    __syncthreads();
    if (s < 31) {   // prefetch next k-slab while this one computes
      aptr += 32; bptr += 32;
      ar0 = *(const v8bf*)aptr; ar1 = *(const v8bf*)(aptr + 8);
      br0 = *(const v8bf*)bptr; br1 = *(const v8bf*)(bptr + 8);
    }
    BF16x16 a0, a1;
    a0.s.lo = *(const v8bf*)(&As[bufoff + a0off + hlf * 8]);
    a0.s.hi = *(const v8bf*)(&As[bufoff + a0off + 16 + hlf * 8]);
    a1.s.lo = *(const v8bf*)(&As[bufoff + a1off + hlf * 8]);
    a1.s.hi = *(const v8bf*)(&As[bufoff + a1off + 16 + hlf * 8]);
#pragma unroll
    for (int ct = 0; ct < 4; ++ct) {
      BF16x16 bf;
      bf.s.lo = *(const v8bf*)(&Bs[bufoff + bboff + ct * 16 * LDS_P + hlf * 8]);
      bf.s.hi = *(const v8bf*)(&Bs[bufoff + bboff + ct * 16 * LDS_P + 16 + hlf * 8]);
      acc[0][ct] = wmma_bf16(a0.v, bf.v, acc[0][ct]);
      acc[1][ct] = wmma_bf16(a1.v, bf.v, acc[1][ct]);
    }
  }
  // epilogue: C row = r + 8*hlf, col = lane%16 (ISA C/D layout)
#pragma unroll
  for (int u = 0; u < 2; ++u)
#pragma unroll
    for (int ct = 0; ct < 4; ++ct) {
      const int col = bn + wn * 64 + ct * 16 + lm;
      const float bv = bias[col];
#pragma unroll
      for (int r = 0; r < 8; ++r) {
        const int row = bm + wm * 32 + u * 16 + r + hlf * 8;
        C[(size_t)row * N + col] = acc[u][ct][r] + bv;
      }
    }
}

// ---------------------------------------------------------------------------
// Implicit-GEMM 1D conv (tap-extended K), divisor-free (shift/mask) indexing.
//   A: bf16 [BSZ][NTOK][K] (K = 1<<kshift), Wt: bf16 [t][o][i], SiLU epilogue.
// Block tile 64x128, 8 waves, wave tile 16x64.
// ---------------------------------------------------------------------------
__global__ __launch_bounds__(256)
void k_gemm_conv(const __bf16* __restrict__ A, const __bf16* __restrict__ Wt,
                 const float* __restrict__ bias, float* __restrict__ C,
                 __bf16* __restrict__ C2, int N, int kshift, int T, int pad) {
  __shared__ __bf16 As[64 * LDS_P];
  __shared__ __bf16 Bs[128 * LDS_P];
  const int kmask = (1 << kshift) - 1;
  const int tid = threadIdx.x;
  const int wave = tid >> 5, lane = tid & 31;
  const int hlf = lane >> 4, lm = lane & 15;
  const int wm = wave >> 1, wn = wave & 1;
  const int bm = blockIdx.y * 64, bn = blockIdx.x * 128;

  v8f acc[4];
#pragma unroll
  for (int ct = 0; ct < 4; ++ct)
#pragma unroll
    for (int r = 0; r < 8; ++r) acc[ct][r] = 0.0f;

  v8bf zero8;
#pragma unroll
  for (int z = 0; z < 8; ++z) zero8[z] = (__bf16)0.0f;

  const int ksteps = (T << kshift) >> 5;
  const int arow = tid >> 2, ak8 = (tid & 3) * 8;
  const int brow = tid >> 1, bk16 = (tid & 1) * 16;
  // row decomposition is static per thread
  const int m = bm + arow;
  const int bb = m >> 10, nn = m & (NTOK - 1);

#pragma unroll 1
  for (int s = 0; s < ksteps; ++s) {
    const int kk = s << 5;
    const int t = kk >> kshift, k0 = kk & kmask;
    {  // stage A with per-tap row shift, zero pad at batch edges
      const int nsrc = nn + t - pad;
      v8bf val = zero8;
      if (nsrc >= 0 && nsrc < NTOK) {
        const __bf16* ap = A + (((size_t)(bb << 10) + nsrc) << kshift) + k0 + ak8;
        val = *(const v8bf*)ap;
      }
      *(v8bf*)(&As[arow * LDS_P + ak8]) = val;
    }
    {  // stage B, Wt is [t][n][k] K-contiguous
      const __bf16* bp = Wt + (((size_t)t * N + (bn + brow)) << kshift) + k0 + bk16;
      *(v8bf*)(&Bs[brow * LDS_P + bk16])     = *(const v8bf*)(bp);
      *(v8bf*)(&Bs[brow * LDS_P + bk16 + 8]) = *(const v8bf*)(bp + 8);
    }
    __syncthreads();
    BF16x16 af;
    af.s.lo = *(const v8bf*)(&As[(wm * 16 + lm) * LDS_P + hlf * 8]);
    af.s.hi = *(const v8bf*)(&As[(wm * 16 + lm) * LDS_P + 16 + hlf * 8]);
#pragma unroll
    for (int ct = 0; ct < 4; ++ct) {
      const int nb = wn * 64 + ct * 16 + lm;
      BF16x16 bf;
      bf.s.lo = *(const v8bf*)(&Bs[nb * LDS_P + hlf * 8]);
      bf.s.hi = *(const v8bf*)(&Bs[nb * LDS_P + 16 + hlf * 8]);
      acc[ct] = wmma_bf16(af.v, bf.v, acc[ct]);
    }
    __syncthreads();
  }
#pragma unroll
  for (int ct = 0; ct < 4; ++ct) {
    const int col = bn + wn * 64 + ct * 16 + lm;
    const float bv = bias[col];
#pragma unroll
    for (int r = 0; r < 8; ++r) {
      const int row = bm + wm * 16 + r + hlf * 8;
      float v = acc[ct][r] + bv;
      v = v / (1.0f + __expf(-v));  // SiLU
      if (C)  C[(size_t)row * N + col] = v;
      if (C2) C2[(size_t)row * N + col] = (__bf16)v;
    }
  }
}

// ---------------------------------------------------------------------------
// conv3 (1x1, 256 -> 1) + bias => patch_logits ; window max
// ---------------------------------------------------------------------------
__global__ __launch_bounds__(256)
void k_conv3(const float* __restrict__ h2, const float* __restrict__ w3,
             const float* __restrict__ b3, float* __restrict__ pl) {
  __shared__ float r8[8];
  int row = blockIdx.x, tid = threadIdx.x;
  float v = h2[(size_t)row * 256 + tid] * w3[tid];
#pragma unroll
  for (int m = 1; m <= 16; m <<= 1) v += __shfl_xor(v, m, 32);
  if ((tid & 31) == 0) r8[tid >> 5] = v;
  __syncthreads();
  if (tid == 0) {
    float s = b3[0];
    for (int w = 0; w < 8; ++w) s += r8[w];
    pl[row] = s;
  }
}

__global__ __launch_bounds__(256)
void k_winmax(const float* __restrict__ pl, float* __restrict__ out) {
  __shared__ float r8[8];
  int b = blockIdx.x, tid = threadIdx.x;
  float m = -1e30f;
#pragma unroll
  for (int i = 0; i < 4; ++i) m = fmaxf(m, pl[b * NTOK + tid + i * 256]);
#pragma unroll
  for (int k = 1; k <= 16; k <<= 1) m = fmaxf(m, __shfl_xor(m, k, 32));
  if ((tid & 31) == 0) r8[tid >> 5] = m;
  __syncthreads();
  if (tid == 0) {
    float mm = -1e30f;
    for (int w = 0; w < 8; ++w) mm = fmaxf(mm, r8[w]);
    out[(size_t)MROWS * DMODEL + b] = mm;
  }
}

// ---------------------------------------------------------------------------
// zero-centered RMSNorm -> bf16
// ---------------------------------------------------------------------------
__global__ __launch_bounds__(256)
void k_norm(const float* __restrict__ x, const float* __restrict__ g,
            __bf16* __restrict__ hn) {
  __shared__ float red[16];
  int row = blockIdx.x, tid = threadIdx.x;
  const float4 v = ((const float4*)(x + (size_t)row * DMODEL))[tid];
  float s1 = v.x + v.y + v.z + v.w;
  float s2 = v.x * v.x + v.y * v.y + v.z * v.z + v.w * v.w;
#pragma unroll
  for (int m = 1; m <= 16; m <<= 1) { s1 += __shfl_xor(s1, m, 32); s2 += __shfl_xor(s2, m, 32); }
  if ((tid & 31) == 0) { red[tid >> 5] = s1; red[8 + (tid >> 5)] = s2; }
  __syncthreads();
  float t1 = 0.f, t2 = 0.f;
  for (int w = 0; w < 8; ++w) { t1 += red[w]; t2 += red[8 + w]; }
  float mean = t1 * (1.0f / DMODEL);
  float ms = t2 * (1.0f / DMODEL) - mean * mean;
  float rstd = rsqrtf(ms + 1e-8f);
#pragma unroll
  for (int i = 0; i < 4; ++i) {
    int d = tid * 4 + i;
    float xv = (i == 0 ? v.x : i == 1 ? v.y : i == 2 ? v.z : v.w);
    hn[(size_t)row * DMODEL + d] = (__bf16)((xv - mean) * rstd * g[d]);
  }
}

// ---------------------------------------------------------------------------
// RoPE in place on qkv buffer: pairs (2j,2j+1), angle = n * theta^(-j/32)
// ---------------------------------------------------------------------------
__global__ __launch_bounds__(512)
void k_rope(float* __restrict__ qkv) {
  int bn = blockIdx.x;
  int n = bn & (NTOK - 1);
  int t = threadIdx.x;          // H*32 = 512 tasks
  int h = t >> 5, j = t & 31;
  float ang = (float)n * __expf(-(float)j * (9.210340371976184f / 32.0f));
  float c, s;
  __sincosf(ang, &s, &c);
  float* q = qkv + (size_t)bn * (3 * DMODEL) + h * DH + 2 * j;
  float* k = q + DMODEL;
  float qr = q[0], qi = q[1];
  q[0] = qr * c - qi * s; q[1] = qr * s + qi * c;
  float kr = k[0], ki = k[1];
  k[0] = kr * c - ki * s; k[1] = kr * s + ki * c;
}

// ---------------------------------------------------------------------------
// Flash attention, bf16 WMMA. Block = 4 waves = 64 queries of one (b,h).
// grid (NTOK/64, NH, BSZ).  Writes bf16 output directly (feeds out-proj GEMM).
// ---------------------------------------------------------------------------
__global__ __launch_bounds__(128)
void k_attn(const float* __restrict__ qkv, const float* __restrict__ pl,
            __bf16* __restrict__ yb) {
  __shared__ __bf16 Qs[64 * 72];
  __shared__ __bf16 Ks[32 * 72];
  __shared__ __bf16 Vs[64 * 40];     // Vs[d][key]
  __shared__ __bf16 Ps[4 * 16 * 40]; // per-wave P scratch
  const int qblk = blockIdx.x, h = blockIdx.y, b = blockIdx.z;
  const int tid = threadIdx.x;
  const int wave = tid >> 5, lane = tid & 31;
  const int hlf = lane >> 4, lm = lane & 15;
  const int qbase = qblk * 64;
  const float scale = 0.125f;   // 1/sqrt(64)

  { // stage Q (64 x 64) once, f32 -> bf16
    int row = tid >> 1, doff = (tid & 1) * 32;
    const float* src = qkv + ((size_t)(b * NTOK + qbase + row)) * (3 * DMODEL) + h * DH + doff;
#pragma unroll
    for (int j = 0; j < 32; ++j) Qs[row * 72 + doff + j] = (__bf16)src[j];
  }

  v8f Y[4];
#pragma unroll
  for (int dt = 0; dt < 4; ++dt)
#pragma unroll
    for (int r = 0; r < 8; ++r) Y[dt][r] = 0.0f;
  float mrow[8], lrow[8];
#pragma unroll
  for (int r = 0; r < 8; ++r) { mrow[r] = -1e30f; lrow[r] = 0.0f; }

#pragma unroll 1
  for (int kb = 0; kb < NTOK / 32; ++kb) {
    __syncthreads();
    { // stage K (32 x 64) and V transposed (Vs[d][key])
      int row = tid >> 2, doff = (tid & 3) * 16;
      const float* ks = qkv + ((size_t)(b * NTOK + kb * 32 + row)) * (3 * DMODEL) + DMODEL + h * DH + doff;
      const float* vs = ks + DMODEL;
#pragma unroll
      for (int j = 0; j < 16; ++j) {
        Ks[row * 72 + doff + j] = (__bf16)ks[j];
        Vs[(doff + j) * 40 + row] = (__bf16)vs[j];
      }
    }
    __syncthreads();

    // S = Q Kt : two 16x16 col tiles, K-dim = 64 (2 wmma steps each)
    v8f s0, s1;
#pragma unroll
    for (int r = 0; r < 8; ++r) { s0[r] = 0.0f; s1[r] = 0.0f; }
#pragma unroll
    for (int ks2 = 0; ks2 < 2; ++ks2) {
      BF16x16 a;
      a.s.lo = *(const v8bf*)(&Qs[(wave * 16 + lm) * 72 + ks2 * 32 + hlf * 16]);
      a.s.hi = *(const v8bf*)(&Qs[(wave * 16 + lm) * 72 + ks2 * 32 + hlf * 16 + 8]);
      BF16x16 b0, b1;
      b0.s.lo = *(const v8bf*)(&Ks[lm * 72 + ks2 * 32 + hlf * 16]);
      b0.s.hi = *(const v8bf*)(&Ks[lm * 72 + ks2 * 32 + hlf * 16 + 8]);
      b1.s.lo = *(const v8bf*)(&Ks[(16 + lm) * 72 + ks2 * 32 + hlf * 16]);
      b1.s.hi = *(const v8bf*)(&Ks[(16 + lm) * 72 + ks2 * 32 + hlf * 16 + 8]);
      s0 = wmma_bf16(a.v, b0.v, s0);
      s1 = wmma_bf16(a.v, b1.v, s1);
    }
    // bias + scale + online softmax
    float pl0 = pl[b * NTOK + kb * 32 + lm];
    float pl1 = pl[b * NTOK + kb * 32 + 16 + lm];
    float fac[8], p0[8], p1[8];
#pragma unroll
    for (int r = 0; r < 8; ++r) {
      float a0 = s0[r] * scale + pl0;
      float a1 = s1[r] * scale + pl1;
      s0[r] = a0; s1[r] = a1;
      float nm = fmaxf(mrow[r], redmax16(fmaxf(a0, a1)));
      fac[r] = __expf(mrow[r] - nm);
      mrow[r] = nm;
    }
#pragma unroll
    for (int r = 0; r < 8; ++r) {
      p0[r] = __expf(s0[r] - mrow[r]);
      p1[r] = __expf(s1[r] - mrow[r]);
      lrow[r] = lrow[r] * fac[r] + redsum16(p0[r] + p1[r]);
    }
#pragma unroll
    for (int dt = 0; dt < 4; ++dt)
#pragma unroll
      for (int r = 0; r < 8; ++r) Y[dt][r] *= fac[r];
    // P (C-layout) -> LDS -> A-layout, then Y += P @ V
#pragma unroll
    for (int r = 0; r < 8; ++r) {
      Ps[(wave * 16 + r + 8 * hlf) * 40 + lm] = (__bf16)p0[r];
      Ps[(wave * 16 + r + 8 * hlf) * 40 + 16 + lm] = (__bf16)p1[r];
    }
    asm volatile("s_wait_dscnt 0x0" ::: "memory");
    BF16x16 pa;
    pa.s.lo = *(const v8bf*)(&Ps[(wave * 16 + lm) * 40 + hlf * 8]);
    pa.s.hi = *(const v8bf*)(&Ps[(wave * 16 + lm) * 40 + 16 + hlf * 8]);
#pragma unroll
    for (int dt = 0; dt < 4; ++dt) {
      BF16x16 vb;
      vb.s.lo = *(const v8bf*)(&Vs[(dt * 16 + lm) * 40 + hlf * 8]);
      vb.s.hi = *(const v8bf*)(&Vs[(dt * 16 + lm) * 40 + 16 + hlf * 8]);
      Y[dt] = wmma_bf16(pa.v, vb.v, Y[dt]);
    }
  }
  // finalize: divide by l, write bf16 y[b][n][h*64+d]
#pragma unroll
  for (int dt = 0; dt < 4; ++dt)
#pragma unroll
    for (int r = 0; r < 8; ++r) {
      int row = qbase + wave * 16 + r + 8 * hlf;
      int d = dt * 16 + lm;
      yb[((size_t)(b * NTOK + row)) * DMODEL + h * DH + d] = (__bf16)(Y[dt][r] / lrow[r]);
    }
}

// ---------------------------------------------------------------------------
// gated residual: x += sigmoid(g) * o
// ---------------------------------------------------------------------------
__global__ __launch_bounds__(256)
void k_residual(float* __restrict__ x, const float* __restrict__ o,
                const float* __restrict__ g) {
  int base = blockIdx.x * 1024 + threadIdx.x * 4;
#pragma unroll
  for (int i = 0; i < 4; ++i) {
    float gv = 1.0f / (1.0f + __expf(-g[base + i]));
    x[base + i] += gv * o[base + i];
  }
}

// ---------------------------------------------------------------------------
// Host launcher
// ---------------------------------------------------------------------------
extern "C" void kernel_launch(void* const* d_in, const int* in_sizes, int n_in,
                              void* d_out, int out_size, void* d_ws, size_t ws_size,
                              hipStream_t stream) {
  const float* patches = (const float*)d_in[0];
  const float* embed_w = (const float*)d_in[1];
  const float* embed_b = (const float*)d_in[2];
  const float* bd_w1   = (const float*)d_in[3];
  const float* bd_b1   = (const float*)d_in[4];
  const float* bd_w2   = (const float*)d_in[5];
  const float* bd_b2   = (const float*)d_in[6];
  const float* bd_w3   = (const float*)d_in[7];
  const float* bd_b3   = (const float*)d_in[8];
  const float* norm_g  = (const float*)d_in[9];
  const float* qkv_w   = (const float*)d_in[10];
  const float* qkv_b   = (const float*)d_in[11];
  const float* out_w   = (const float*)d_in[12];
  const float* out_b   = (const float*)d_in[13];
  const float* gate_w  = (const float*)d_in[14];
  const float* gate_b  = (const float*)d_in[15];
  float* xout = (float*)d_out;                 // x lives in d_out[0..4M)

  // workspace carve-up
  char* wsb = (char*)d_ws;
  size_t off = 0;
  auto carve = [&](size_t bytes) { void* p = wsb + off; off = (off + bytes + 255) & ~(size_t)255; return p; };
  __bf16* xh  = (__bf16*)carve((size_t)MROWS * DMODEL * 2);
  __bf16* hn  = (__bf16*)carve((size_t)MROWS * DMODEL * 2);
  float*  qkv = (float*) carve((size_t)MROWS * 3 * DMODEL * 4);
  __bf16* yb  = (__bf16*)carve((size_t)MROWS * DMODEL * 2);
  float*  obuf= (float*) carve((size_t)MROWS * DMODEL * 4);
  float*  gbuf= (float*) carve((size_t)MROWS * DMODEL * 4);
  __bf16* h1b = (__bf16*)carve((size_t)MROWS * 512 * 2);
  float*  h2  = (float*) carve((size_t)MROWS * 256 * 4);
  float*  pl  = (float*) carve((size_t)MROWS * 4);
  __bf16* wq  = (__bf16*)carve((size_t)3 * 3072 * 1024 * 2);
  __bf16* wo  = (__bf16*)carve((size_t)3 * 1024 * 1024 * 2);
  __bf16* wg  = (__bf16*)carve((size_t)3 * 1024 * 1024 * 2);
  __bf16* w1  = (__bf16*)carve((size_t)5 * 512 * 1024 * 2);
  __bf16* w2  = (__bf16*)carve((size_t)3 * 256 * 512 * 2);
  (void)ws_size; (void)out_size; (void)n_in; (void)in_sizes;

  // ---- weight repacks (bf16, B-fragment friendly layouts)
  for (int l = 0; l < 3; ++l) {
    k_transpose_cast<<<(1024 * 3072 + 255) / 256, 256, 0, stream>>>(
        qkv_w + (size_t)l * 1024 * 3072, wq + (size_t)l * 3072 * 1024, 1024, 3072);
    k_transpose_cast<<<(1024 * 1024 + 255) / 256, 256, 0, stream>>>(
        out_w + (size_t)l * 1024 * 1024, wo + (size_t)l * 1024 * 1024, 1024, 1024);
    k_transpose_cast<<<(1024 * 1024 + 255) / 256, 256, 0, stream>>>(
        gate_w + (size_t)l * 1024 * 1024, wg + (size_t)l * 1024 * 1024, 1024, 1024);
  }
  k_repack_conv<<<(5 * 512 * 1024 + 255) / 256, 256, 0, stream>>>(bd_w1, w1, 512, 1024, 5);
  k_repack_conv<<<(3 * 256 * 512 + 255) / 256, 256, 0, stream>>>(bd_w2, w2, 256, 512, 3);

  // ---- patch embed (+ bf16 mirror)
  k_embed<<<MROWS, 256, 0, stream>>>(patches, embed_w, embed_b, xout, xh);

  // ---- boundary detector: conv1 (K=1024,T=5) -> silu bf16; conv2 -> silu f32
  k_gemm_conv<<<dim3(512 / 128, MROWS / 64), 256, 0, stream>>>(
      xh, w1, bd_b1, nullptr, h1b, 512, 10, 5, 2);
  k_gemm_conv<<<dim3(256 / 128, MROWS / 64), 256, 0, stream>>>(
      h1b, w2, bd_b2, h2, nullptr, 256, 9, 3, 1);
  k_conv3<<<MROWS, 256, 0, stream>>>(h2, bd_w3, bd_b3, pl);
  k_winmax<<<BSZ, 256, 0, stream>>>(pl, xout);

  // ---- transformer layers
  for (int l = 0; l < 3; ++l) {
    k_norm<<<MROWS, 256, 0, stream>>>(xout, norm_g + (size_t)l * DMODEL, hn);
    k_gemm_dense<<<dim3(3072 / 128, MROWS / 128), 256, 0, stream>>>(
        hn, wq + (size_t)l * 3072 * 1024, qkv_b + (size_t)l * 3072, qkv, 3072);
    k_rope<<<MROWS, 512, 0, stream>>>(qkv);
    k_attn<<<dim3(NTOK / 64, NH, BSZ), 128, 0, stream>>>(qkv, pl, yb);
    k_gemm_dense<<<dim3(1024 / 128, MROWS / 128), 256, 0, stream>>>(
        yb, wo + (size_t)l * 1024 * 1024, out_b + (size_t)l * 1024, obuf, 1024);
    k_gemm_dense<<<dim3(1024 / 128, MROWS / 128), 256, 0, stream>>>(
        hn, wg + (size_t)l * 1024 * 1024, gate_b + (size_t)l * 1024, gbuf, 1024);
    k_residual<<<MROWS, 256, 0, stream>>>(xout, obuf, gbuf);
  }
}
